// SoftSplatBaseline_39788577030945
// MI455X (gfx1250) — matrix operand order repaired and checked by reading, not compiled, not run on hardware
//
#include <hip/hip_runtime.h>

// ---------------------------------------------------------------------------
// Types
// ---------------------------------------------------------------------------
typedef __attribute__((ext_vector_type(16))) __bf16 v16bf;
typedef __attribute__((ext_vector_type(8)))  __bf16 v8bf;
typedef __attribute__((ext_vector_type(8)))  float  v8f;
typedef __attribute__((ext_vector_type(4)))  int    v4i;

// ---------------------------------------------------------------------------
// Async global->LDS copy (CDNA5 GLOBAL_LOAD_ASYNC_TO_LDS_B128, ASYNCcnt).
// Params: (int4 AS1*, int4 AS3*, imm offset, imm cpol) — probe-verified.
// ---------------------------------------------------------------------------
#if defined(__HIP_DEVICE_COMPILE__) && __has_builtin(__builtin_amdgcn_global_load_async_to_lds_b128)
#define HAS_ASYNC_LDS 1
typedef v4i __attribute__((address_space(1))) as1_v4i;
typedef v4i __attribute__((address_space(3))) as3_v4i;
#else
#define HAS_ASYNC_LDS 0
#endif

__device__ __forceinline__ void async_b128(const __bf16* g, __bf16* l) {
#if HAS_ASYNC_LDS
  __builtin_amdgcn_global_load_async_to_lds_b128(
      (as1_v4i*)(unsigned long long)g,
      (as3_v4i*)(unsigned int)(unsigned long long)l, 0, 0);
#else
  *(v8bf*)l = *(const v8bf*)g;
#endif
}

template <int N>
__device__ __forceinline__ void async_wait() {
#if HAS_ASYNC_LDS
#if __has_builtin(__builtin_amdgcn_s_wait_asynccnt)
  __builtin_amdgcn_s_wait_asynccnt(N);
#else
  asm volatile("s_wait_asynccnt %0" :: "i"(N) : "memory");
#endif
#endif
}

// ---------------------------------------------------------------------------
// Weight packing: OIHW fp32 -> B-fragment blocks of 512 bf16.
// Fragment f = (ky*3+kx)*KC + kc, tile nt. Element [lane][i] = W[o][ci][ky][kx]
// with n = lane&15, k = (lane>>4)*16 + i, o = nt*16+n, ci = kc*32+k
// (contiguous k order for wmma_f32_16x16x32_bf16, the same order the conv A loads use).
// ---------------------------------------------------------------------------
__global__ void ss_pack_weights_kernel(const float* __restrict__ w, __bf16* __restrict__ wp,
                                       int CinReal, int CoutReal, int KC, int NT, int total) {
  int idx = blockIdx.x * blockDim.x + threadIdx.x;
  if (idx >= total) return;
  int i = idx & 15;
  int lane = (idx >> 4) & 31;
  int frag = idx >> 9;
  int nt = frag % NT;
  int f  = frag / NT;
  int kc = f % KC;
  int tap = f / KC;
  int ky = tap / 3, kx = tap % 3;
  int n = lane & 15;
  int k = ((lane >> 4) * 16) + i;
  int o  = nt * 16 + n;
  int ci = kc * 32 + k;
  float v = 0.0f;
  if (o < CoutReal && ci < CinReal)
    v = w[(((long long)o * CinReal + ci) * 3 + ky) * 3 + kx];
  wp[idx] = (__bf16)v;
}

// ---------------------------------------------------------------------------
// Standalone PReLU pass (bf16, 8 elements/thread). Applying PReLU once per
// element here is 9x cheaper in ALU than fusing it into the conv A-loads
// (each input element is read by 9 taps); the extra bf16 round-trip is
// negligible at 23.3 TB/s.
// ---------------------------------------------------------------------------
__global__ void ss_prelu_kernel(const __bf16* __restrict__ in, const float* __restrict__ slope_p,
                                __bf16* out, long long n8) {
  long long idx = (long long)blockIdx.x * blockDim.x + threadIdx.x;
  if (idx >= n8) return;
  const float s = slope_p[0];
  v8bf v = *(const v8bf*)(in + idx * 8);
  v8bf o;
#pragma unroll
  for (int i = 0; i < 8; ++i) {
    float f = (float)v[i];
    f = (f >= 0.f) ? f : f * s;
    o[i] = (__bf16)f;
  }
  *(v8bf*)(out + idx * 8) = o;
}

// ---------------------------------------------------------------------------
// Implicit-GEMM 3x3 conv (pad=1), NHWC bf16 activations, f32 WMMA accum.
// One wave => one 16x16 output tile (16 pixels along W x 16 output channels).
// Templated on KC = Cin/32: fully unrolled ky*kx*kc reduction, 9*KC static
// v_wmma per instantiation, B fragments double-buffered via ASYNCcnt.
// A fragments load straight into WMMA registers (union), no repacking.
// Fused epilogue: bias, residual add, output PReLU.
// ---------------------------------------------------------------------------
__device__ __forceinline__ void stage_frag(const __bf16* g, __bf16* lbase, int lane) {
  async_b128(g + lane * 8,       lbase + lane * 8);
  async_b128(g + lane * 8 + 256, lbase + lane * 8 + 256);
}

union AFrag { v16bf v; v8bf h[2]; };

template <int KC>
__global__ __launch_bounds__(256)
void ss_conv3x3_wmma_kernel(const __bf16* __restrict__ in, const __bf16* __restrict__ wp,
                            const float* __restrict__ bias,
                            const float* __restrict__ out_slope_p, const __bf16* res,
                            __bf16* out,
                            int B, int Hi, int Wi, int Ho, int Wo,
                            int CoutReal, int stride, int NT) {
  __shared__ __align__(64) __bf16 ldsw[8][2][512];
  const int CinBuf = KC * 32;
  const int wv   = threadIdx.x >> 5;
  const int lane = threadIdx.x & 31;
  const int wtiles = (Wo + 15) >> 4;
  const long long total = (long long)B * Ho * wtiles * NT;
  long long tile = (long long)blockIdx.x * 8 + wv;
  if (tile >= total) return;
  int nt = (int)(tile % NT); long long t = tile / NT;
  int wt = (int)(t % wtiles); t /= wtiles;
  int oh = (int)(t % Ho);
  int b  = (int)(t / Ho);

  const int m  = lane & 15;
  const int hiHalf = lane >> 4;          // 0 or 1
  const int klo = hiHalf * 16;           // K base of this lane half's 16 elements
  const int ow = wt * 16 + m;
  const bool mvalid = (ow < Wo);

  v8f acc;
#pragma unroll
  for (int r = 0; r < 8; ++r) acc[r] = 0.0f;
  const v8bf zero8 = {};

  const size_t wstride = (size_t)NT * 512;        // elements between fragments
  const __bf16* wcur = wp + (size_t)nt * 512;     // fragment 0 for this tile
  const __bf16* wstage = wcur + wstride;          // fragment to prefetch next
  stage_frag(wcur, &ldsw[wv][0][0], lane);

#pragma unroll
  for (int ky = 0; ky < 3; ++ky) {
    const int ih = oh * stride + ky - 1;
    const bool hval = (ih >= 0) & (ih < Hi);
    const __bf16* rowp = in + ((long long)(b * Hi + ih) * Wi) * CinBuf + klo;
#pragma unroll
    for (int kx = 0; kx < 3; ++kx) {
      const int iw = ow * stride + kx - 1;
      const bool pval = mvalid & hval & (iw >= 0) & (iw < Wi);
      const __bf16* px = rowp + (long long)iw * CinBuf;
#pragma unroll
      for (int kc = 0; kc < KC; ++kc) {
        const int f = (ky * 3 + kx) * KC + kc;   // compile-time after unroll
        if (f + 1 < 9 * KC) {
          stage_frag(wstage, &ldsw[wv][(f + 1) & 1][0], lane);
          wstage += wstride;
          async_wait<2>();   // 2 newest (next frag) may be pending; frag f done
        } else {
          async_wait<0>();
        }
        v16bf bfrag = *(const v16bf*)&ldsw[wv][f & 1][lane * 16];

        AFrag a;
        const __bf16* s = px + kc * 32;
        a.h[0] = pval ? *(const v8bf*)s        : zero8;  // K = klo .. klo+7
        a.h[1] = pval ? *(const v8bf*)(s + 8)  : zero8;  // K = klo+8 .. klo+15
        acc = __builtin_amdgcn_wmma_f32_16x16x32_bf16(false, a.v, false, bfrag,
                                                      (short)0, acc, false, false);
      }
    }
  }

  // Epilogue: lane holds column n = lane&15, rows m = r + hiHalf*8.
  const int cout = nt * 16 + m;
  if (cout >= CoutReal) return;
  const float bv = bias ? bias[cout] : 0.0f;
  const bool do_os = (out_slope_p != nullptr);
  const float oslope = do_os ? out_slope_p[0] : 1.0f;
#pragma unroll
  for (int r = 0; r < 8; ++r) {
    const int mm = r + hiHalf * 8;
    const int ow2 = wt * 16 + mm;
    if (ow2 >= Wo) continue;
    const long long oidx = ((long long)(b * Ho + oh) * Wo + ow2) * CoutReal + cout;
    float v = acc[r] + bv;
    if (res) v += (float)res[oidx];
    if (do_os) v = (v >= 0.f) ? v : v * oslope;
    out[oidx] = (__bf16)v;
  }
}

// ---------------------------------------------------------------------------
// Elementwise kernels
// ---------------------------------------------------------------------------

// x (4,3,2,H,W) fp32 -> fcat/rev (8,H,W,32) bf16, scaled to [-1,1], ch padded.
__global__ void ss_preprocess_kernel(const float* __restrict__ x, __bf16* __restrict__ fcat,
                                     __bf16* __restrict__ rev, int B, int H, int W) {
  long long idx = (long long)blockIdx.x * blockDim.x + threadIdx.x;
  long long tot = (long long)2 * B * H * W;
  if (idx >= tot) return;
  int w = (int)(idx % W); long long t = idx / W;
  int h = (int)(t % H);
  int bb = (int)(t / H);
  int b = bb % B, fr = bb / B;
  long long obase = (((long long)bb * H + h) * W + w) * 32;
  for (int c = 0; c < 32; ++c) {
    float vf = 0.f, vr = 0.f;
    if (c < 3) {
      vf = x[((((long long)b * 3 + c) * 2 + fr)     * H + h) * W + w] * 2.f - 1.f;
      vr = x[((((long long)b * 3 + c) * 2 + (1-fr)) * H + h) * W + w] * 2.f - 1.f;
    }
    fcat[obase + c] = (__bf16)vf;
    rev[obase + c]  = (__bf16)vr;
  }
}

__device__ __forceinline__ float ss_bw_tap(const __bf16* rev, int bb, int H, int W, int c,
                                           int ix, int iy, float wt) {
  bool valid = (ix >= 0) && (ix < W) && (iy >= 0) && (iy < H);
  int cx = min(max(ix, 0), W - 1);
  int cy = min(max(iy, 0), H - 1);
  float v = (float)rev[(((long long)bb * H + cy) * W + cx) * 32 + c];
  return valid ? v * wt : 0.f;
}

// z = alpha * sum_c |backwarp(rev, flow) - fcat|
__global__ void ss_backwarp_absdiff_kernel(const __bf16* __restrict__ fcat,
                                           const __bf16* __restrict__ rev,
                                           const float* __restrict__ flow,
                                           const float* __restrict__ alpha,
                                           float* __restrict__ z, int B2, int H, int W) {
  long long idx = (long long)blockIdx.x * blockDim.x + threadIdx.x;
  long long tot = (long long)B2 * H * W;
  if (idx >= tot) return;
  int w = (int)(idx % W); long long t = idx / W;
  int h = (int)(t % H);
  int bb = (int)(t / H);
  long long hw = (long long)H * W;
  float fx = flow[((long long)bb * 2 + 0) * hw + (long long)h * W + w];
  float fy = flow[((long long)bb * 2 + 1) * hw + (long long)h * W + w];
  float px = (w + fx) * (float)(W - 1) / (float)W;
  float py = (h + fy) * (float)(H - 1) / (float)H;
  float x0f = floorf(px), y0f = floorf(py);
  float wx1 = px - x0f, wy1 = py - y0f;
  int x0 = (int)x0f, y0 = (int)y0f;
  float s = 0.f;
  for (int c = 0; c < 3; ++c) {
    float v = ss_bw_tap(rev, bb, H, W, c, x0,     y0,     (1.f-wx1)*(1.f-wy1))
            + ss_bw_tap(rev, bb, H, W, c, x0 + 1, y0,     wx1*(1.f-wy1))
            + ss_bw_tap(rev, bb, H, W, c, x0,     y0 + 1, (1.f-wx1)*wy1)
            + ss_bw_tap(rev, bb, H, W, c, x0 + 1, y0 + 1, wx1*wy1);
    s += fabsf(v - (float)fcat[(((long long)bb * H + h) * W + w) * 32 + c]);
  }
  z[(long long)bb * hw + (long long)h * W + w] = alpha[0] * s;
}

// fp32 NCHW bilinear resize (reference sampling), with output scale.
__global__ void ss_resize_f32_kernel(const float* __restrict__ src, float* __restrict__ dst,
                                     int B, int C, int Hs, int Ws, int Hd, int Wd, float scale) {
  long long idx = (long long)blockIdx.x * blockDim.x + threadIdx.x;
  long long tot = (long long)B * C * Hd * Wd;
  if (idx >= tot) return;
  int ox = (int)(idx % Wd); long long t = idx / Wd;
  int oy = (int)(t % Hd); t /= Hd;
  int c = (int)(t % C);
  int b = (int)(t / C);
  float sy = fmaxf((oy + 0.5f) * (float)Hs / (float)Hd - 0.5f, 0.f);
  float sx = fmaxf((ox + 0.5f) * (float)Ws / (float)Wd - 0.5f, 0.f);
  int y0 = (int)floorf(sy); float wy = sy - y0; int y1 = min(y0 + 1, Hs - 1);
  int x0 = (int)floorf(sx); float wx = sx - x0; int x1 = min(x0 + 1, Ws - 1);
  const float* p = src + ((long long)b * C + c) * Hs * Ws;
  float a00 = p[(long long)y0 * Ws + x0], a10 = p[(long long)y1 * Ws + x0];
  float a01 = p[(long long)y0 * Ws + x1], a11 = p[(long long)y1 * Ws + x1];
  float r0 = a00 + (a10 - a00) * wy;
  float r1 = a01 + (a11 - a01) * wy;
  dst[idx] = (r0 + (r1 - r0) * wx) * scale;
}

// flowt = concat(flow_lv[:4]*t, flow_lv[4:]*(1-t))
__global__ void ss_flow_t_kernel(const float* __restrict__ fl, const float* __restrict__ tt,
                                 float* __restrict__ out, int B, long long HW2) {
  long long idx = (long long)blockIdx.x * blockDim.x + threadIdx.x;
  long long tot = (long long)2 * B * HW2;
  if (idx >= tot) return;
  int bb = (int)(idx / HW2);
  float f = (bb < B) ? tt[bb] : 1.f - tt[bb - B];
  out[idx] = fl[idx] * f;
}

// Forward splat with exp(z) weighting into (B2,H,W,C+1) fp32 accumulator.
__global__ void ss_softsplat_accum_kernel(const __bf16* __restrict__ feat,
                                          const float* __restrict__ flowt,
                                          const float* __restrict__ z,
                                          float* __restrict__ acc,
                                          int B2, int H, int W, int C) {
  long long idx = (long long)blockIdx.x * blockDim.x + threadIdx.x;
  long long tot = (long long)B2 * H * W;
  if (idx >= tot) return;
  int w = (int)(idx % W); long long t = idx / W;
  int h = (int)(t % H);
  int bb = (int)(t / H);
  long long hw = (long long)H * W;
  long long pix = (long long)bb * hw + (long long)h * W + w;
  float e = __expf(z[pix]);
  float gx = w + flowt[((long long)bb * 2 + 0) * hw + (long long)h * W + w];
  float gy = h + flowt[((long long)bb * 2 + 1) * hw + (long long)h * W + w];
  float x0f = floorf(gx), y0f = floorf(gy);
  float wx1 = gx - x0f, wy1 = gy - y0f;
  int x0 = (int)x0f, y0 = (int)y0f;
  int ix[4] = {x0, x0 + 1, x0, x0 + 1};
  int iy[4] = {y0, y0, y0 + 1, y0 + 1};
  float wt[4] = {(1.f-wx1)*(1.f-wy1), wx1*(1.f-wy1), (1.f-wx1)*wy1, wx1*wy1};
  const __bf16* fp = feat + pix * C;
  for (int c = 0; c <= C; ++c) {
    float v = (c < C) ? (float)fp[c] : 1.0f;
    v *= e;
    for (int k = 0; k < 4; ++k) {
      int xx = ix[k], yy = iy[k];
      if (xx >= 0 && xx < W && yy >= 0 && yy < H)
        atomicAdd(&acc[(((long long)bb * H + yy) * W + xx) * (C + 1) + c], v * wt[k]);
    }
  }
}

// Normalize and concat the two half-batches: out (4,H,W,2C) bf16.
__global__ void ss_softsplat_norm_kernel(const float* __restrict__ acc, __bf16* __restrict__ out,
                                         int B, int H, int W, int C) {
  long long idx = (long long)blockIdx.x * blockDim.x + threadIdx.x;
  long long tot = (long long)B * H * W * 2 * C;
  if (idx >= tot) return;
  int c2 = (int)(idx % (2 * C)); long long t = idx / (2 * C);
  int w = (int)(t % W); t /= W;
  int h = (int)(t % H);
  int b = (int)(t / H);
  int bb = b + ((c2 < C) ? 0 : B);
  int c = (c2 < C) ? c2 : c2 - C;
  const float* a = acc + (((long long)bb * H + h) * W + w) * (C + 1);
  float den = a[C];
  den = (den == 0.f) ? 1.f : den;
  out[idx] = (__bf16)(a[c] / den);
}

// bf16 NHWC bilinear resize (used by UB 2x upsample).
__global__ void ss_resize_feat_kernel(const __bf16* __restrict__ src, __bf16* __restrict__ dst,
                                      int B, int C, int Hs, int Ws, int Hd, int Wd) {
  long long idx = (long long)blockIdx.x * blockDim.x + threadIdx.x;
  long long tot = (long long)B * Hd * Wd * C;
  if (idx >= tot) return;
  int c = (int)(idx % C); long long t = idx / C;
  int ox = (int)(t % Wd); t /= Wd;
  int oy = (int)(t % Hd);
  int b = (int)(t / Hd);
  float sy = fmaxf((oy + 0.5f) * (float)Hs / (float)Hd - 0.5f, 0.f);
  float sx = fmaxf((ox + 0.5f) * (float)Ws / (float)Wd - 0.5f, 0.f);
  int y0 = (int)floorf(sy); float wy = sy - y0; int y1 = min(y0 + 1, Hs - 1);
  int x0 = (int)floorf(sx); float wx = sx - x0; int x1 = min(x0 + 1, Ws - 1);
  const __bf16* p = src + (long long)b * Hs * Ws * C + c;
  float a00 = (float)p[((long long)y0 * Ws + x0) * C];
  float a10 = (float)p[((long long)y1 * Ws + x0) * C];
  float a01 = (float)p[((long long)y0 * Ws + x1) * C];
  float a11 = (float)p[((long long)y1 * Ws + x1) * C];
  float r0 = a00 + (a10 - a00) * wy;
  float r1 = a01 + (a11 - a01) * wy;
  dst[idx] = (__bf16)(r0 + (r1 - r0) * wx);
}

// rgb (4,H,W,3) bf16 -> d_out (4,3,H,W) fp32, clip((v+1)*0.5, 0, 1)
__global__ void ss_finalize_kernel(const __bf16* __restrict__ rgb, float* __restrict__ out,
                                   int B, int H, int W) {
  long long idx = (long long)blockIdx.x * blockDim.x + threadIdx.x;
  long long tot = (long long)B * 3 * H * W;
  if (idx >= tot) return;
  int w = (int)(idx % W); long long t = idx / W;
  int h = (int)(t % H); t /= H;
  int c = (int)(t % 3);
  int b = (int)(t / 3);
  float v = (float)rgb[(((long long)b * H + h) * W + w) * 3 + c];
  v = (v + 1.f) * 0.5f;
  out[idx] = fminf(fmaxf(v, 0.f), 1.f);
}

// ---------------------------------------------------------------------------
// Host orchestration
// ---------------------------------------------------------------------------
namespace {

struct Bump { char* base; size_t off; };
static inline void* bump_alloc(Bump& a, size_t bytes) {
  void* r = a.base + a.off;
  a.off += (bytes + 255) & ~(size_t)255;
  return r;
}
static inline unsigned cdiv(long long a, long long b) { return (unsigned)((a + b - 1) / b); }

struct Blk { const float *a1, *a2, *c1b, *c1w, *c2b, *c2w; };

static void run_conv(hipStream_t s, Bump& ws,
                     const __bf16* in, const float* w, const float* bias,
                     const float* oslope, const __bf16* res,
                     __bf16* out, int B, int Hi, int Wi, int stride,
                     int CinBuf, int CinReal, int CoutReal) {
  int Ho = (Hi + stride - 1) / stride;
  int Wo = (Wi + stride - 1) / stride;
  int KC = CinBuf / 32;
  int NT = (CoutReal + 15) / 16;
  size_t wpe = (size_t)9 * KC * NT * 512;
  __bf16* wp = (__bf16*)bump_alloc(ws, wpe * sizeof(__bf16));
  ss_pack_weights_kernel<<<cdiv((long long)wpe, 256), 256, 0, s>>>(
      w, wp, CinReal, CoutReal, KC, NT, (int)wpe);
  long long tiles = (long long)B * Ho * ((Wo + 15) / 16) * NT;
  unsigned blocks = cdiv(tiles, 8);
  switch (KC) {
    case 1: ss_conv3x3_wmma_kernel<1><<<blocks, 256, 0, s>>>(
        in, wp, bias, oslope, res, out, B, Hi, Wi, Ho, Wo, CoutReal, stride, NT); break;
    case 2: ss_conv3x3_wmma_kernel<2><<<blocks, 256, 0, s>>>(
        in, wp, bias, oslope, res, out, B, Hi, Wi, Ho, Wo, CoutReal, stride, NT); break;
    case 3: ss_conv3x3_wmma_kernel<3><<<blocks, 256, 0, s>>>(
        in, wp, bias, oslope, res, out, B, Hi, Wi, Ho, Wo, CoutReal, stride, NT); break;
    case 4: ss_conv3x3_wmma_kernel<4><<<blocks, 256, 0, s>>>(
        in, wp, bias, oslope, res, out, B, Hi, Wi, Ho, Wo, CoutReal, stride, NT); break;
    case 6: ss_conv3x3_wmma_kernel<6><<<blocks, 256, 0, s>>>(
        in, wp, bias, oslope, res, out, B, Hi, Wi, Ho, Wo, CoutReal, stride, NT); break;
    default: break;  // channel counts in this network are always 32..192
  }
}

static void run_prelu(hipStream_t s, const __bf16* in, const float* slope, __bf16* out,
                      long long nElems) {
  long long n8 = nElems / 8;  // all feature tensors have C % 8 == 0
  ss_prelu_kernel<<<cdiv(n8, 256), 256, 0, s>>>(in, slope, out, n8);
}

} // namespace

extern "C" void kernel_launch(void* const* d_in, const int* in_sizes, int n_in,
                              void* d_out, int out_size, void* d_ws, size_t ws_size,
                              hipStream_t stream) {
  (void)in_sizes; (void)n_in; (void)out_size; (void)ws_size;
  const int B4 = 4, B8 = 8;
  const int H0 = 256, W0 = 448;
  const int H1 = 128, W1 = 224;
  const int H2 = 64,  W2 = 112;
  const long long n0 = (long long)H0 * W0, n1 = (long long)H1 * W1, n2 = (long long)H2 * W2;

  const float* x    = (const float*)d_in[0];
  const float* tt   = (const float*)d_in[1];
  const float* flow = (const float*)d_in[2];
  // params: dict keys alphabetical, lists in order. Leaf layout:
  //  0: alpha | 1..6: compress (b,w)x3 | 7..42: down[2][3] blks
  //  43..60: fp[3] blks | 61..156: lateral (6,5,5) blks
  //  157..159: to_rgb (a, c.b, c.w) | 160..195: up[2][3] blks
  auto getf = [&](int leaf) -> const float* { return (const float*)d_in[3 + leaf]; };
  auto blk = [&](int base) -> Blk {
    return { getf(base), getf(base + 1), getf(base + 2), getf(base + 3),
             getf(base + 4), getf(base + 5) };
  };

  Bump ws{ (char*)d_ws, 0 };  // assumes ws_size >= ~1.4 GB

  // ---- preprocess: scaled & channel-padded bf16 NHWC frames --------------
  __bf16* fcat = (__bf16*)bump_alloc(ws, (size_t)B8 * n0 * 32 * 2);
  __bf16* rev  = (__bf16*)bump_alloc(ws, (size_t)B8 * n0 * 32 * 2);
  ss_preprocess_kernel<<<cdiv((long long)B8 * n0, 256), 256, 0, stream>>>(x, fcat, rev, B4, H0, W0);

  // ---- feature pyramid (batch 8) -----------------------------------------
  __bf16* t0   = (__bf16*)bump_alloc(ws, (size_t)B8 * n0 * 32 * 2);
  __bf16* pyr0 = (__bf16*)bump_alloc(ws, (size_t)B8 * n0 * 32 * 2);
  __bf16* t1   = (__bf16*)bump_alloc(ws, (size_t)B8 * n1 * 64 * 2);
  __bf16* pyr1 = (__bf16*)bump_alloc(ws, (size_t)B8 * n1 * 64 * 2);
  __bf16* t2   = (__bf16*)bump_alloc(ws, (size_t)B8 * n2 * 96 * 2);
  __bf16* pyr2 = (__bf16*)bump_alloc(ws, (size_t)B8 * n2 * 96 * 2);
  {
    Blk f0 = blk(43), f1 = blk(49), f2 = blk(55);
    run_conv(stream, ws, fcat, f0.c1w, f0.c1b, f0.a1, nullptr, t0,   B8, H0, W0, 1, 32, 3,  32);
    run_conv(stream, ws, t0,   f0.c2w, f0.c2b, f0.a2, nullptr, pyr0, B8, H0, W0, 1, 32, 32, 32);
    run_conv(stream, ws, pyr0, f1.c1w, f1.c1b, f1.a1, nullptr, t1,   B8, H0, W0, 2, 32, 32, 64);
    run_conv(stream, ws, t1,   f1.c2w, f1.c2b, f1.a2, nullptr, pyr1, B8, H1, W1, 1, 64, 64, 64);
    run_conv(stream, ws, pyr1, f2.c1w, f2.c1b, f2.a1, nullptr, t2,   B8, H1, W1, 2, 64, 64, 96);
    run_conv(stream, ws, t2,   f2.c2w, f2.c2b, f2.a2, nullptr, pyr2, B8, H2, W2, 1, 96, 96, 96);
  }

  // ---- z = alpha * sum|backwarp(rev) - fcat| ------------------------------
  float* z = (float*)bump_alloc(ws, (size_t)B8 * n0 * 4);
  ss_backwarp_absdiff_kernel<<<cdiv((long long)B8 * n0, 256), 256, 0, stream>>>(
      fcat, rev, flow, getf(0), z, B8, H0, W0);

  // ---- per-level softsplat ------------------------------------------------
  const __bf16* pyrs[3] = { pyr0, pyr1, pyr2 };
  const int Cl[3] = { 32, 64, 96 };
  const int Hl[3] = { H0, H1, H2 };
  const int Wl[3] = { W0, W1, W2 };
  __bf16* warped[3];
  for (int lv = 0; lv < 3; ++lv) {
    const long long nl = (long long)Hl[lv] * Wl[lv];
    const int C = Cl[lv];
    float scale = 1.0f / (float)(1 << lv);
    float* flow_lv = (float*)bump_alloc(ws, (size_t)B8 * 2 * nl * 4);
    ss_resize_f32_kernel<<<cdiv((long long)B8 * 2 * nl, 256), 256, 0, stream>>>(
        flow, flow_lv, B8, 2, H0, W0, Hl[lv], Wl[lv], scale);
    float* flowt = (float*)bump_alloc(ws, (size_t)B8 * 2 * nl * 4);
    ss_flow_t_kernel<<<cdiv((long long)B8 * 2 * nl, 256), 256, 0, stream>>>(
        flow_lv, tt, flowt, B4, 2 * nl);
    float* z_lv = (float*)bump_alloc(ws, (size_t)B8 * nl * 4);
    ss_resize_f32_kernel<<<cdiv((long long)B8 * nl, 256), 256, 0, stream>>>(
        z, z_lv, B8, 1, H0, W0, Hl[lv], Wl[lv], 1.0f);
    size_t accBytes = (size_t)B8 * nl * (C + 1) * 4;
    float* acc = (float*)bump_alloc(ws, accBytes);
    (void)hipMemsetAsync(acc, 0, accBytes, stream);
    ss_softsplat_accum_kernel<<<cdiv((long long)B8 * nl, 256), 256, 0, stream>>>(
        pyrs[lv], flowt, z_lv, acc, B8, Hl[lv], Wl[lv], C);
    warped[lv] = (__bf16*)bump_alloc(ws, (size_t)B4 * nl * 2 * C * 2);
    ss_softsplat_norm_kernel<<<cdiv((long long)B4 * nl * 2 * C, 256), 256, 0, stream>>>(
        acc, warped[lv], B4, Hl[lv], Wl[lv], C);
  }

  // ---- gridnet ------------------------------------------------------------
  __bf16 *x0[7], *x1[6], *x2[6];
  for (int i = 0; i < 7; ++i) x0[i] = (__bf16*)bump_alloc(ws, (size_t)B4 * n0 * 32 * 2);
  for (int i = 0; i < 6; ++i) x1[i] = (__bf16*)bump_alloc(ws, (size_t)B4 * n1 * 64 * 2);
  for (int i = 0; i < 6; ++i) x2[i] = (__bf16*)bump_alloc(ws, (size_t)B4 * n2 * 96 * 2);
  __bf16* r0_tmp = (__bf16*)bump_alloc(ws, (size_t)B4 * n0 * 32 * 2);
  __bf16* r0_rs  = (__bf16*)bump_alloc(ws, (size_t)B4 * n0 * 64 * 2);
  __bf16* r1_tmp = (__bf16*)bump_alloc(ws, (size_t)B4 * n1 * 64 * 2);
  __bf16* r1_rs  = (__bf16*)bump_alloc(ws, (size_t)B4 * n1 * 96 * 2);
  __bf16* r2_tmp = (__bf16*)bump_alloc(ws, (size_t)B4 * n2 * 96 * 2);
  __bf16* p0 = (__bf16*)bump_alloc(ws, (size_t)B4 * n0 * 32 * 2);  // prelu scratch row0
  __bf16* p1 = (__bf16*)bump_alloc(ws, (size_t)B4 * n1 * 64 * 2);  // prelu scratch row1
  __bf16* p2 = (__bf16*)bump_alloc(ws, (size_t)B4 * n2 * 96 * 2);  // prelu scratch row2

  // LB: xout = xin + c2(prelu_a2_epilogue(c1(prelu_a1(xin))))
  auto LB = [&](Blk q, const __bf16* xin, __bf16* xout, __bf16* tmp, __bf16* pr,
                int H, int W, int C) {
    run_prelu(stream, xin, q.a1, pr, (long long)B4 * H * W * C);
    run_conv(stream, ws, pr,  q.c1w, q.c1b, q.a2, nullptr, tmp,  B4, H, W, 1, C, C, C);
    run_conv(stream, ws, tmp, q.c2w, q.c2b, nullptr, xin, xout, B4, H, W, 1, C, C, C);
  };
  auto DB = [&](Blk q, const __bf16* xin, __bf16* target, __bf16* tmp, __bf16* pr,
                int H, int W, int Cin, int Cout) {
    run_prelu(stream, xin, q.a1, pr, (long long)B4 * H * W * Cin);
    run_conv(stream, ws, pr,  q.c1w, q.c1b, q.a2, nullptr, tmp, B4, H, W, 2, Cin, Cin, Cout);
    run_conv(stream, ws, tmp, q.c2w, q.c2b, nullptr, target, target,
             B4, H / 2, W / 2, 1, Cout, Cout, Cout);
  };
  auto UB = [&](Blk q, const __bf16* xin, __bf16* target, __bf16* rs, __bf16* tmp,
                int Hs, int Ws, int Cin, int Cout) {
    int Hd = 2 * Hs, Wd = 2 * Ws;
    ss_resize_feat_kernel<<<cdiv((long long)B4 * Hd * Wd * Cin, 256), 256, 0, stream>>>(
        xin, rs, B4, Cin, Hs, Ws, Hd, Wd);
    run_prelu(stream, rs, q.a1, rs, (long long)B4 * Hd * Wd * Cin);  // in-place
    run_conv(stream, ws, rs,  q.c1w, q.c1b, q.a2, nullptr, tmp, B4, Hd, Wd, 1, Cin, Cin, Cout);
    run_conv(stream, ws, tmp, q.c2w, q.c2b, nullptr, target, target,
             B4, Hd, Wd, 1, Cout, Cout, Cout);
  };

  // compress
  run_conv(stream, ws, warped[0], getf(2), getf(1), nullptr, nullptr, x0[0],
           B4, H0, W0, 1, 64, 64, 32);
  run_conv(stream, ws, warped[1], getf(4), getf(3), nullptr, nullptr, x1[0],
           B4, H1, W1, 1, 128, 128, 64);
  run_conv(stream, ws, warped[2], getf(6), getf(5), nullptr, nullptr, x2[0],
           B4, H2, W2, 1, 192, 192, 96);

  LB(blk(61), x0[0], x0[1], r0_tmp, p0, H0, W0, 32);              // x01
  LB(blk(67), x0[1], x0[2], r0_tmp, p0, H0, W0, 32);              // x02
  LB(blk(73), x0[2], x0[3], r0_tmp, p0, H0, W0, 32);              // x03
  DB(blk(7),  x0[0], x1[0], r1_tmp, p0, H0, W0, 32, 64);          // x10 += DB(x00)
  DB(blk(25), x1[0], x2[0], r2_tmp, p1, H1, W1, 64, 96);          // x20 += DB(x10)
  LB(blk(97),  x1[0], x1[1], r1_tmp, p1, H1, W1, 64);             // x11
  LB(blk(127), x2[0], x2[1], r2_tmp, p2, H2, W2, 96);             // x21
  DB(blk(13), x0[1], x1[1], r1_tmp, p0, H0, W0, 32, 64);          // x11 += DB(x01)
  DB(blk(31), x1[1], x2[1], r2_tmp, p1, H1, W1, 64, 96);          // x21 += DB(x11)
  LB(blk(103), x1[1], x1[2], r1_tmp, p1, H1, W1, 64);             // x12
  LB(blk(133), x2[1], x2[2], r2_tmp, p2, H2, W2, 96);             // x22
  DB(blk(19), x0[2], x1[2], r1_tmp, p0, H0, W0, 32, 64);          // x12 += DB(x02)
  DB(blk(37), x1[2], x2[2], r2_tmp, p1, H1, W1, 64, 96);          // x22 += DB(x12)
  LB(blk(109), x1[2], x1[3], r1_tmp, p1, H1, W1, 64);             // x13
  LB(blk(139), x2[2], x2[3], r2_tmp, p2, H2, W2, 96);             // x23
  LB(blk(145), x2[3], x2[4], r2_tmp, p2, H2, W2, 96);             // x24
  LB(blk(151), x2[4], x2[5], r2_tmp, p2, H2, W2, 96);             // x25
  UB(blk(178), x2[3], x1[3], r1_rs, r1_tmp, H2, W2, 96, 64);      // x13 += UB(x23)
  UB(blk(160), x1[3], x0[3], r0_rs, r0_tmp, H1, W1, 64, 32);      // x03 += UB(x13)
  LB(blk(115), x1[3], x1[4], r1_tmp, p1, H1, W1, 64);             // x14
  LB(blk(79),  x0[3], x0[4], r0_tmp, p0, H0, W0, 32);             // x04
  UB(blk(184), x2[4], x1[4], r1_rs, r1_tmp, H2, W2, 96, 64);      // x14 += UB(x24)
  UB(blk(166), x1[4], x0[4], r0_rs, r0_tmp, H1, W1, 64, 32);      // x04 += UB(x14)
  LB(blk(121), x1[4], x1[5], r1_tmp, p1, H1, W1, 64);             // x15
  LB(blk(85),  x0[4], x0[5], r0_tmp, p0, H0, W0, 32);             // x05
  UB(blk(190), x2[5], x1[5], r1_rs, r1_tmp, H2, W2, 96, 64);      // x15 += UB(x25)
  UB(blk(172), x1[5], x0[5], r0_rs, r0_tmp, H1, W1, 64, 32);      // x05 += UB(x15)
  LB(blk(91), x0[5], x0[6], r0_tmp, p0, H0, W0, 32);              // outF

  // to_rgb: conv(prelu(outF, a), c)
  __bf16* rgb = (__bf16*)bump_alloc(ws, (size_t)B4 * n0 * 3 * 2);
  run_prelu(stream, x0[6], getf(157), p0, (long long)B4 * n0 * 32);
  run_conv(stream, ws, p0, getf(159), getf(158), nullptr, nullptr, rgb,
           B4, H0, W0, 1, 32, 32, 3);
  ss_finalize_kernel<<<cdiv((long long)B4 * 3 * n0, 256), 256, 0, stream>>>(
      rgb, (float*)d_out, B4, H0, W0);
}
